// CQAttention_80444737454823
// MI455X (gfx1250) — compile-verified, matrix operand order checked by reading, not run on hardware
//
#include <hip/hip_runtime.h>
#include <hip/hip_bf16.h>

#define K_B  16
#define K_D  128
#define K_LC 2048
#define K_LQ 512
#define NEGV (-1e30f)

typedef __attribute__((ext_vector_type(16))) __bf16 v16bf;
typedef __attribute__((ext_vector_type(8)))  __bf16 v8bf;
typedef __attribute__((ext_vector_type(4)))  __bf16 v4bf;
typedef __attribute__((ext_vector_type(8)))  float  v8f;
typedef __attribute__((ext_vector_type(4)))  float  v4f;

#define LDT 40  // tile leading dim (bf16 elems): 80B rows -> 16B-aligned b128 frag loads

__device__ __forceinline__ v8f wmma_bf16(v16bf a, v16bf b, v8f c) {
  // D = A(16x32 bf16) * B(32x16 bf16) + C(f32)
  return __builtin_amdgcn_wmma_f32_16x16x32_bf16(false, a, false, b, (short)0, c, false, false);
}

__device__ __forceinline__ v8f zero8() {
  v8f z = {0.f, 0.f, 0.f, 0.f, 0.f, 0.f, 0.f, 0.f};
  return z;
}

// Fragment load for a row-major 16x32 tile (rows = M for A-operand, rows = N for
// B-operand stored transposed [n][k]).  Per ISA 7.12.2 the per-lane data is two
// contiguous 8-element (16B) runs of row (lane&15):  [8h, 8h+8) and [16+8h, 24+8h)
// => exactly two ds_load_b128.
__device__ __forceinline__ v16bf load_frag_row(const __bf16* row0) {
  int lane = threadIdx.x & 31;
  const __bf16* p = row0 + (lane & 15) * LDT + 8 * (lane >> 4);
  v8bf lo = *(const v8bf*)(p);
  v8bf hi = *(const v8bf*)(p + 16);
  return __builtin_shufflevector(lo, hi, 0, 1, 2, 3, 4, 5, 6, 7,
                                 8, 9, 10, 11, 12, 13, 14, 15);
}

// ---------------- k0: rank-1 terms sub0[b][c], sub1[b][q] ----------------
__global__ __launch_bounds__(256) void k0_rowdots(
    const float* __restrict__ C, const float* __restrict__ Q,
    const float* __restrict__ w4C, const float* __restrict__ w4Q,
    float* __restrict__ sub0, float* __restrict__ sub1) {
  int b = blockIdx.z;
  int i = blockIdx.x * 256 + threadIdx.x;
  if (blockIdx.y == 0) {
    const float* Cb = C + (size_t)b * K_D * K_LC;
    float s = 0.f;
#pragma unroll 4
    for (int d = 0; d < K_D; ++d) s += Cb[(size_t)d * K_LC + i] * w4C[d];
    sub0[b * K_LC + i] = s;
  } else {
    if (i >= K_LQ) return;
    const float* Qb = Q + (size_t)b * K_D * K_LQ;
    float s = 0.f;
#pragma unroll 4
    for (int d = 0; d < K_D; ++d) s += Qb[(size_t)d * K_LQ + i] * w4Q[d];
    sub1[b * K_LQ + i] = s;
  }
}

// ---------------- k1: S = (Ct*w4mlu) @ Qt^T + sub0 + sub1 + bias ----------
// Block tile 128(c) x 64(q), 8 waves as 4x2, wave tile 32x32, K over D=128.
__global__ __launch_bounds__(256) void k1_similarity(
    const float* __restrict__ C, const float* __restrict__ Q,
    const float* __restrict__ w4mlu, const float* __restrict__ bias,
    const float* __restrict__ sub0, const float* __restrict__ sub1,
    float* __restrict__ S) {
  __shared__ __bf16 sa[128][LDT];   // A: [c][d-chunk]
  __shared__ __bf16 sbT[64][LDT];   // B transposed: [q][d-chunk]
  int tid = threadIdx.x;
  int b  = blockIdx.z;
  int c0 = blockIdx.x * 128;
  int q0 = blockIdx.y * 64;
  const float* Cb = C + (size_t)b * K_D * K_LC;
  const float* Qb = Q + (size_t)b * K_D * K_LQ;
  int w = tid >> 5, wr = w >> 1, wc = w & 1;

  v8f acc[2][2];
#pragma unroll
  for (int i = 0; i < 2; ++i)
#pragma unroll
    for (int j = 0; j < 2; ++j) acc[i][j] = zero8();

  for (int kc = 0; kc < K_D; kc += 32) {
    __syncthreads();
    // A stage: float4 along c (coalesced global), 4 strided b16 LDS stores
    for (int idx = tid; idx < 32 * 32; idx += 256) {
      int mg = idx & 31, k = idx >> 5;
      int m4 = mg * 4;
      v4f cv = *(const v4f*)(Cb + (size_t)(kc + k) * K_LC + c0 + m4);
      float wm = w4mlu[kc + k];
#pragma unroll
      for (int j = 0; j < 4; ++j) sa[m4 + j][k] = (__bf16)(cv[j] * wm);
    }
    // B stage: float4 along q (coalesced global), 4 strided b16 LDS stores
    for (int idx = tid; idx < 16 * 32; idx += 256) {
      int ng = idx & 15, k = idx >> 4;
      int n4 = ng * 4;
      v4f qv = *(const v4f*)(Qb + (size_t)(kc + k) * K_LQ + q0 + n4);
#pragma unroll
      for (int j = 0; j < 4; ++j) sbT[n4 + j][k] = (__bf16)qv[j];
    }
    __syncthreads();
    v16bf af[2], bf[2];
#pragma unroll
    for (int ti = 0; ti < 2; ++ti) af[ti] = load_frag_row(&sa[wr * 32 + ti * 16][0]);
#pragma unroll
    for (int tj = 0; tj < 2; ++tj) bf[tj] = load_frag_row(&sbT[wc * 32 + tj * 16][0]);
#pragma unroll
    for (int ti = 0; ti < 2; ++ti)
#pragma unroll
      for (int tj = 0; tj < 2; ++tj)
        acc[ti][tj] = wmma_bf16(af[ti], bf[tj], acc[ti][tj]);
  }

  float bv = bias[0];
  int lane = tid & 31, n = lane & 15, h = lane >> 4;
#pragma unroll
  for (int ti = 0; ti < 2; ++ti)
#pragma unroll
    for (int tj = 0; tj < 2; ++tj)
#pragma unroll
      for (int r = 0; r < 8; ++r) {
        int c = c0 + wr * 32 + ti * 16 + r + 8 * h;
        int q = q0 + wc * 32 + tj * 16 + n;
        S[((size_t)b * K_LC + c) * K_LQ + q] =
            acc[ti][tj][r] + sub0[b * K_LC + c] + sub1[b * K_LQ + q] + bv;
      }
}

// ---------------- k2: row softmax stats (over q), one wave per row --------
__global__ __launch_bounds__(256) void k2_rowstats(
    const float* __restrict__ S, const int* __restrict__ Qmask,
    float* __restrict__ rowmax, float* __restrict__ rowinv) {
  int row  = blockIdx.x * 8 + (threadIdx.x >> 5);  // row = b*LC + c
  int lane = threadIdx.x & 31;
  int b = row / K_LC;
  const float* Sr = S + (size_t)row * K_LQ + lane * 16;  // 16 contiguous per lane
  const int* qm = Qmask + b * K_LQ + lane * 16;
  float v[16];
  float mx = NEGV;
#pragma unroll
  for (int g = 0; g < 4; ++g) {
    v4f x = *(const v4f*)(Sr + g * 4);
#pragma unroll
    for (int j = 0; j < 4; ++j) {
      float s = qm[g * 4 + j] ? x[j] : NEGV;
      v[g * 4 + j] = s;
      mx = fmaxf(mx, s);
    }
  }
#pragma unroll
  for (int o = 16; o >= 1; o >>= 1) mx = fmaxf(mx, __shfl_xor(mx, o, 32));
  float sum = 0.f;
#pragma unroll
  for (int j = 0; j < 16; ++j) sum += __expf(v[j] - mx);
#pragma unroll
  for (int o = 16; o >= 1; o >>= 1) sum += __shfl_xor(sum, o, 32);
  if (lane == 0) { rowmax[row] = mx; rowinv[row] = 1.f / sum; }
}

// ---------------- k3: column softmax stats (over c), online ---------------
__global__ __launch_bounds__(256) void k3_colstats(
    const float* __restrict__ S, const int* __restrict__ Cmask,
    float* __restrict__ colmax, float* __restrict__ colinv) {
  int b = blockIdx.z;
  int q = blockIdx.x * 256 + threadIdx.x;
  const float* Sb = S + (size_t)b * K_LC * K_LQ;
  const int* cm = Cmask + b * K_LC;
  float m = NEGV, s = 0.f;
  for (int c = 0; c < K_LC; ++c) {
    if (c + 16 < K_LC)  // stream hint: 2KB-strided walk -> global_prefetch_b8
      __builtin_prefetch(Sb + (size_t)(c + 16) * K_LQ + q, 0, 1);
    float vv = cm[c] ? Sb[(size_t)c * K_LQ + q] : NEGV;
    float nm = fmaxf(m, vv);
    s = s * __expf(m - nm) + __expf(vv - nm);
    m = nm;
  }
  colmax[b * K_LQ + q] = m;
  colinv[b * K_LQ + q] = 1.f / s;
}

// ---------------- k4: T[b][q][d] = sum_c S2[c][q] * Ct[c][d] (bf16 out) ---
// Block tile 64(q) x 128(d), 8 waves as 2x4, K over c = 2048.
__global__ __launch_bounds__(256) void k4_Tmat(
    const float* __restrict__ S, const float* __restrict__ C,
    const int* __restrict__ Cmask,
    const float* __restrict__ colmax, const float* __restrict__ colinv,
    __bf16* __restrict__ T) {
  __shared__ __bf16 sa[64][LDT];    // A: [q][c-chunk]
  __shared__ __bf16 sbT[128][LDT];  // B transposed: [d][c-chunk]
  int tid = threadIdx.x;
  int b  = blockIdx.z;
  int q0 = blockIdx.x * 64;
  const float* Sb = S + (size_t)b * K_LC * K_LQ;
  const float* Cb = C + (size_t)b * K_D * K_LC;
  const int* cm = Cmask + b * K_LC;
  int w = tid >> 5, wr = w >> 2, wc = w & 3;

  v8f acc[2][2];
#pragma unroll
  for (int i = 0; i < 2; ++i)
#pragma unroll
    for (int j = 0; j < 2; ++j) acc[i][j] = zero8();

  for (int c0 = 0; c0 < K_LC; c0 += 32) {
    __syncthreads();
    // A stage: float4 along q, apply col-softmax + mask, 4 strided b16 stores
    for (int idx = tid; idx < 16 * 32; idx += 256) {
      int mg = idx & 15, k = idx >> 4;
      int m4 = mg * 4, c = c0 + k;
      v4f sv = *(const v4f*)(Sb + (size_t)c * K_LQ + q0 + m4);
      float cmv = (float)cm[c];
#pragma unroll
      for (int j = 0; j < 4; ++j) {
        int q = q0 + m4 + j;
        sa[m4 + j][k] =
            (__bf16)(__expf(sv[j] - colmax[b * K_LQ + q]) * colinv[b * K_LQ + q] * cmv);
      }
    }
    // B stage: float4 along c (coalesced) -> bf16x4 contiguous ds_store_b64
    for (int idx = tid; idx < 128 * 8; idx += 256) {
      int kg = idx & 7, n = idx >> 3;
      int k4 = kg * 4;
      v4f cv = *(const v4f*)(Cb + (size_t)n * K_LC + c0 + k4);
      v4bf o;
#pragma unroll
      for (int j = 0; j < 4; ++j) o[j] = (__bf16)cv[j];
      *(v4bf*)(&sbT[n][k4]) = o;
    }
    __syncthreads();
    v16bf af[2], bf[2];
#pragma unroll
    for (int ti = 0; ti < 2; ++ti) af[ti] = load_frag_row(&sa[wr * 32 + ti * 16][0]);
#pragma unroll
    for (int tj = 0; tj < 2; ++tj) bf[tj] = load_frag_row(&sbT[wc * 32 + tj * 16][0]);
#pragma unroll
    for (int ti = 0; ti < 2; ++ti)
#pragma unroll
      for (int tj = 0; tj < 2; ++tj)
        acc[ti][tj] = wmma_bf16(af[ti], bf[tj], acc[ti][tj]);
  }

  int lane = tid & 31, n = lane & 15, h = lane >> 4;
#pragma unroll
  for (int ti = 0; ti < 2; ++ti)
#pragma unroll
    for (int tj = 0; tj < 2; ++tj)
#pragma unroll
      for (int r = 0; r < 8; ++r) {
        int q = q0 + wr * 32 + ti * 16 + r + 8 * h;
        int d = wc * 32 + tj * 16 + n;
        T[((size_t)b * K_LQ + q) * K_D + d] = (__bf16)acc[ti][tj][r];
      }
}

// ---------------- k5: A = S1@Qt, Bmat = S1@T, fused output ----------------
// Block tile 128(c) x 64(d), 8 waves as 4x2, K over q = 512. Shared S1 frags.
union K5Smem {
  struct {
    __bf16 a[128][LDT];   // S1 tile [c][q-chunk]
    __bf16 bq[64][LDT];   // Qt tile transposed [d][q-chunk]
    __bf16 bt[64][LDT];   // T  tile transposed [d][q-chunk]
  } st;
  float epi[64][132];     // [d][c] transpose buffer; 132 -> 16B-aligned rows
};

__global__ __launch_bounds__(256) void k5_output(
    const float* __restrict__ S, const float* __restrict__ C,
    const float* __restrict__ Q, const __bf16* __restrict__ T,
    const int* __restrict__ Qmask,
    const float* __restrict__ rowmax, const float* __restrict__ rowinv,
    float* __restrict__ out) {
  __shared__ K5Smem sm;
  int tid = threadIdx.x;
  int b  = blockIdx.z;
  int c0 = blockIdx.x * 128;
  int d0 = blockIdx.y * 64;
  const float* Sb = S + (size_t)b * K_LC * K_LQ;
  const float* Cb = C + (size_t)b * K_D * K_LC;
  const float* Qb = Q + (size_t)b * K_D * K_LQ;
  const int* qm = Qmask + b * K_LQ;
  int w = tid >> 5, wr = w >> 1, wc = w & 1;

  v8f accA[2][2], accB[2][2];
#pragma unroll
  for (int i = 0; i < 2; ++i)
#pragma unroll
    for (int j = 0; j < 2; ++j) { accA[i][j] = zero8(); accB[i][j] = zero8(); }

  for (int qc = 0; qc < K_LQ; qc += 32) {
    __syncthreads();
    // S1 stage: float4 along q, softmax-normalize, bf16x4 contiguous store
    for (int idx = tid; idx < 128 * 8; idx += 256) {
      int kg = idx & 7, m = idx >> 3;
      int k4 = kg * 4, c = c0 + m;
      v4f sv = *(const v4f*)(Sb + (size_t)c * K_LQ + qc + k4);
      float rm = rowmax[b * K_LC + c], ri = rowinv[b * K_LC + c];
      v4bf o;
#pragma unroll
      for (int j = 0; j < 4; ++j)
        o[j] = (__bf16)(__expf(sv[j] - rm) * ri * (float)qm[qc + k4 + j]);
      *(v4bf*)(&sm.st.a[m][k4]) = o;
    }
    // Qt stage: float4 along q (coalesced) -> bf16x4 contiguous store
    for (int idx = tid; idx < 64 * 8; idx += 256) {
      int kg = idx & 7, n = idx >> 3;
      int k4 = kg * 4;
      v4f qv = *(const v4f*)(Qb + (size_t)(d0 + n) * K_LQ + qc + k4);
      v4bf o;
#pragma unroll
      for (int j = 0; j < 4; ++j) o[j] = (__bf16)qv[j];
      *(v4bf*)(&sm.st.bq[n][k4]) = o;
    }
    // T stage: bf16x4 along d (coalesced), 4 strided b16 stores
    for (int idx = tid; idx < 16 * 32; idx += 256) {
      int ng = idx & 15, k = idx >> 4;
      int n4 = ng * 4;
      v4bf tv = *(const v4bf*)(T + ((size_t)b * K_LQ + qc + k) * K_D + d0 + n4);
#pragma unroll
      for (int j = 0; j < 4; ++j) sm.st.bt[n4 + j][k] = tv[j];
    }
    __syncthreads();
    v16bf af[2], bfq[2], bft[2];
#pragma unroll
    for (int ti = 0; ti < 2; ++ti) af[ti] = load_frag_row(&sm.st.a[wr * 32 + ti * 16][0]);
#pragma unroll
    for (int tj = 0; tj < 2; ++tj) {
      bfq[tj] = load_frag_row(&sm.st.bq[wc * 32 + tj * 16][0]);
      bft[tj] = load_frag_row(&sm.st.bt[wc * 32 + tj * 16][0]);
    }
#pragma unroll
    for (int ti = 0; ti < 2; ++ti)
#pragma unroll
      for (int tj = 0; tj < 2; ++tj) {
        accA[ti][tj] = wmma_bf16(af[ti], bfq[tj], accA[ti][tj]);
        accB[ti][tj] = wmma_bf16(af[ti], bft[tj], accB[ti][tj]);
      }
  }

  int lane = tid & 31, nn = lane & 15, h = lane >> 4;
  size_t ob = (size_t)b * 4 * K_D * K_LC;

  // Pass 1: A -> out blocks 0 (Ct copy), 1 (A), 2 (Ct*A); float4 stores over c
  __syncthreads();
#pragma unroll
  for (int ti = 0; ti < 2; ++ti)
#pragma unroll
    for (int tj = 0; tj < 2; ++tj)
#pragma unroll
      for (int r = 0; r < 8; ++r)
        sm.epi[wc * 32 + tj * 16 + nn][wr * 32 + ti * 16 + r + 8 * h] = accA[ti][tj][r];
  __syncthreads();
  for (int idx = tid; idx < 64 * 32; idx += 256) {
    int c4 = (idx & 31) * 4, dd = idx >> 5;
    int gc = c0 + c4, gd = d0 + dd;
    v4f cv = *(const v4f*)(Cb + (size_t)gd * K_LC + gc);
    v4f av = *(const v4f*)(&sm.epi[dd][c4]);
    *(v4f*)(out + ob + (size_t)gd * K_LC + gc)             = cv;
    *(v4f*)(out + ob + (size_t)(K_D + gd) * K_LC + gc)     = av;
    *(v4f*)(out + ob + (size_t)(2 * K_D + gd) * K_LC + gc) = cv * av;
  }

  // Pass 2: Bmat -> out block 3 (Ct*Bmat)
  __syncthreads();
#pragma unroll
  for (int ti = 0; ti < 2; ++ti)
#pragma unroll
    for (int tj = 0; tj < 2; ++tj)
#pragma unroll
      for (int r = 0; r < 8; ++r)
        sm.epi[wc * 32 + tj * 16 + nn][wr * 32 + ti * 16 + r + 8 * h] = accB[ti][tj][r];
  __syncthreads();
  for (int idx = tid; idx < 64 * 32; idx += 256) {
    int c4 = (idx & 31) * 4, dd = idx >> 5;
    int gc = c0 + c4, gd = d0 + dd;
    v4f cv = *(const v4f*)(Cb + (size_t)gd * K_LC + gc);
    v4f bv = *(const v4f*)(&sm.epi[dd][c4]);
    *(v4f*)(out + ob + (size_t)(3 * K_D + gd) * K_LC + gc) = cv * bv;
  }
}

// --------------------------------------------------------------------------
extern "C" void kernel_launch(void* const* d_in, const int* in_sizes, int n_in,
                              void* d_out, int out_size, void* d_ws, size_t ws_size,
                              hipStream_t stream) {
  const float* C     = (const float*)d_in[0];
  const float* Q     = (const float*)d_in[1];
  const int*   Cmask = (const int*)d_in[2];
  const int*   Qmask = (const int*)d_in[3];
  const float* w4C   = (const float*)d_in[4];
  const float* w4Q   = (const float*)d_in[5];
  const float* w4mlu = (const float*)d_in[6];
  const float* bias  = (const float*)d_in[7];
  float* out = (float*)d_out;

  char* ws = (char*)d_ws;
  size_t offS      = 0;                                        // B*LC*LQ f32
  size_t offSub0   = offS    + (size_t)K_B * K_LC * K_LQ * 4;  // B*LC f32
  size_t offSub1   = offSub0 + (size_t)K_B * K_LC * 4;         // B*LQ f32
  size_t offRowMax = offSub1 + (size_t)K_B * K_LQ * 4;         // B*LC f32
  size_t offRowInv = offRowMax + (size_t)K_B * K_LC * 4;       // B*LC f32
  size_t offColMax = offRowInv + (size_t)K_B * K_LC * 4;       // B*LQ f32
  size_t offColInv = offColMax + (size_t)K_B * K_LQ * 4;       // B*LQ f32
  size_t offT      = offColInv + (size_t)K_B * K_LQ * 4;       // B*LQ*D bf16

  float*  Sws    = (float*)(ws + offS);
  float*  sub0   = (float*)(ws + offSub0);
  float*  sub1   = (float*)(ws + offSub1);
  float*  rowmax = (float*)(ws + offRowMax);
  float*  rowinv = (float*)(ws + offRowInv);
  float*  colmax = (float*)(ws + offColMax);
  float*  colinv = (float*)(ws + offColInv);
  __bf16* Tws    = (__bf16*)(ws + offT);

  dim3 blk(256);
  k0_rowdots <<<dim3(K_LC / 256, 2, K_B), blk, 0, stream>>>(C, Q, w4C, w4Q, sub0, sub1);
  k1_similarity<<<dim3(K_LC / 128, K_LQ / 64, K_B), blk, 0, stream>>>(
      C, Q, w4mlu, bias, sub0, sub1, Sws);
  k2_rowstats<<<dim3((K_B * K_LC) / 8), blk, 0, stream>>>(Sws, Qmask, rowmax, rowinv);
  k3_colstats<<<dim3(K_LQ / 256, 1, K_B), blk, 0, stream>>>(Sws, Cmask, colmax, colinv);
  k4_Tmat    <<<dim3(K_LQ / 64, 1, K_B), blk, 0, stream>>>(
      Sws, C, Cmask, colmax, colinv, Tws);
  k5_output  <<<dim3(K_LC / 128, K_D / 64, K_B), blk, 0, stream>>>(
      Sws, C, Q, Tws, Qmask, rowmax, rowinv, out);
  (void)in_sizes; (void)n_in; (void)out_size; (void)ws_size;
}
